// GAT_63891933495702
// MI455X (gfx1250) — matrix operand order repaired and checked by reading, not compiled, not run on hardware
//
#include <hip/hip_runtime.h>
#include <hip/hip_bf16.h>
#include <stdint.h>

// ---- problem constants (match reference) ----
#define N_NODES 50000
#define IN_F    256
#define HID_    16
#define HEADS_  4
#define T_      2
#define E_      800000
#define EP_     (E_ + N_NODES)          // edges + self loops = 850000
#define OUT_    16
#define HD_     64                      // HEADS_*HID_
#define KTOT    (T_ * N_NODES * HD_)    // 6,400,000  (flat dim of final GEMV)

typedef _Float16 half16 __attribute__((ext_vector_type(16)));
typedef float    float8 __attribute__((ext_vector_type(8)));
typedef unsigned int u32x4 __attribute__((ext_vector_type(4)));
typedef int      i32x4 __attribute__((vector_size(16)));   // matches builtin's V4i pointee

// ---- CDNA5 async global->LDS staging (guarded; falls back to plain copy) ----
#if defined(__has_builtin)
#if __has_builtin(__builtin_amdgcn_global_load_async_to_lds_b128)
#define HAVE_ASYNC_LDS 1
#endif
#endif

#ifdef HAVE_ASYNC_LDS
#define AS1_ __attribute__((address_space(1)))
#define AS3_ __attribute__((address_space(3)))
// generic "shared" pointers carry the LDS byte offset in their low 32 bits (ISA 10.2)
#define ASYNC_COPY16(lp, gp)                                                     \
    __builtin_amdgcn_global_load_async_to_lds_b128(                              \
        (AS1_ i32x4*)(uintptr_t)(gp), (AS3_ i32x4*)(uint32_t)(uintptr_t)(lp), 0, 0)
#if __has_builtin(__builtin_amdgcn_s_wait_asynccnt)
#define ASYNC_WAIT() __builtin_amdgcn_s_wait_asynccnt(0)
#else
#define ASYNC_WAIT() asm volatile("s_wait_asynccnt 0" ::: "memory")
#endif
#else
#define ASYNC_COPY16(lp, gp) (*(u32x4*)(lp) = *(const u32x4*)(gp))
#define ASYNC_WAIT() ((void)0)
#endif

// ---------------- init: rst=bias, emax=-inf, denom=0, out=b_out ----------------
__global__ void gat_init_kernel(float* __restrict__ rst, float* __restrict__ emax,
                                float* __restrict__ denom, float* __restrict__ out,
                                const float* __restrict__ bias,
                                const float* __restrict__ bout) {
  int gid = blockIdx.x * blockDim.x + threadIdx.x;
  if (gid < KTOT) {
    int t = gid / (N_NODES * HD_);
    int f = gid & (HD_ - 1);
    rst[gid] = bias[t * HD_ + f];
  }
  if (gid < T_ * N_NODES * HEADS_) {
    emax[gid]  = -3.0e38f;
    denom[gid] = 0.0f;
  }
  if (gid < OUT_) out[gid] = bout[gid];
}

// ---------------- fp32 -> fp16 conversions ----------------
__global__ void cvt_x_kernel(const float* __restrict__ x, _Float16* __restrict__ xf16) {
  int gid = blockIdx.x * blockDim.x + threadIdx.x;
  if (gid < N_NODES * IN_F) xf16[gid] = (_Float16)x[gid];
}

// W[t][k][n] (T, 256, 64) -> wt16[t][n][k] (n-major, k contiguous) for B fragments
__global__ void cvt_w_kernel(const float* __restrict__ W, _Float16* __restrict__ wt16) {
  int gid = blockIdx.x * blockDim.x + threadIdx.x;
  if (gid >= T_ * HD_ * IN_F) return;
  int t = gid / (HD_ * IN_F);
  int r = gid - t * (HD_ * IN_F);
  int n = r / IN_F;
  int k = r - n * IN_F;
  wt16[gid] = (_Float16)W[((size_t)t * IN_F + k) * HD_ + n];
}

// ---------------- WMMA GEMM: h[t] = X @ W[t]  (M=50000, K=256, N=64) ----------------
// block = 256 threads (8 waves). Each wave owns a 16x64 output tile -> block does 128 rows.
#define AROW 40   // padded LDS row stride (halves): 80B, conflict-free for 16-row frags
__global__ void gemm_wmma_kernel(const _Float16* __restrict__ xf16,
                                 const _Float16* __restrict__ wt16,
                                 float* __restrict__ hout, int t) {
  __shared__ __align__(16) _Float16 sA[128 * AROW];  // 10 KB
  __shared__ __align__(16) _Float16 sB[64 * AROW];   // 5 KB

  const int tid  = threadIdx.x;
  const int wave = tid >> 5;
  const int lane = tid & 31;
  const int mBase = blockIdx.x * 128;
  const _Float16* wb = wt16 + (size_t)t * HD_ * IN_F;

  float8 c[4] = {};

  for (int k0 = 0; k0 < IN_F; k0 += 32) {
    __syncthreads();  // previous-iteration fragment reads done before restaging
    // stage A: 128 rows x 32 halves = 512 chunks of 16B
    #pragma unroll
    for (int i = 0; i < 2; ++i) {
      int cidx = tid + i * 256;
      int row  = cidx >> 2;
      int off  = (cidx & 3) * 8;                 // halves
      int grow = mBase + row;
      if (grow >= N_NODES) grow = N_NODES - 1;   // clamp; masked at store
      ASYNC_COPY16(&sA[row * AROW + off], xf16 + (size_t)grow * IN_F + k0 + off);
    }
    // stage B: 64 rows x 32 halves = 256 chunks of 16B
    {
      int n   = tid >> 2;
      int off = (tid & 3) * 8;
      ASYNC_COPY16(&sB[n * AROW + off], wb + (size_t)n * IN_F + k0 + off);
    }
    ASYNC_WAIT();
    __syncthreads();

    // A fragment (ISA 7.12.2, 16-bit A 16x32): lanes 0-15 K=0..7,16..23; 16-31 K=8..15,24..31
    const int mloc = (wave << 4) + (lane & 15);
    const int k0l  = (lane >= 16) ? 8 : 0;
    union { half16 h; u32x4 q[2]; } a;
    a.q[0] = *(const u32x4*)&sA[mloc * AROW + k0l];
    a.q[1] = *(const u32x4*)&sA[mloc * AROW + 16 + k0l];

    #pragma unroll
    for (int ct = 0; ct < 4; ++ct) {
      // B fragment (32x16): lane holds column n=lane%16, K=0..7,16..23 (lanes<16) / 8..15,24..31
      const int nloc = (ct << 4) + (lane & 15);
      const int kb   = (lane >= 16) ? 8 : 0;
      union { half16 h; u32x4 q[2]; } b;
      b.q[0] = *(const u32x4*)&sB[nloc * AROW + kb];
      b.q[1] = *(const u32x4*)&sB[nloc * AROW + kb + 16];
      c[ct] = __builtin_amdgcn_wmma_f32_16x16x32_f16(
          /*neg_a=*/false, a.h, /*neg_b=*/false, b.h,
          /*c_mod=*/(short)0, c[ct], /*reuse_a=*/false, /*reuse_b=*/false);
    }
  }

  // store C: VGPR r holds M=r (lanes 0-15) / M=r+8 (lanes 16-31); N = lane%16
  const int colBase = lane & 15;
  const int rowOff  = (lane >= 16) ? 8 : 0;
  float* hb = hout + (size_t)t * N_NODES * HD_;
  #pragma unroll
  for (int ct = 0; ct < 4; ++ct)
    #pragma unroll
    for (int r = 0; r < 8; ++r) {
      int gm = mBase + (wave << 4) + rowOff + r;
      if (gm < N_NODES) hb[(size_t)gm * HD_ + (ct << 4) + colBase] = c[ct][r];
    }
}

// ---------------- el/er: per (node, head) dot of h with attn vectors ----------------
__global__ void elr_kernel(const float* __restrict__ h,
                           const float* __restrict__ attn_l,
                           const float* __restrict__ attn_r,
                           float* __restrict__ el, float* __restrict__ er, int t) {
  int gid = blockIdx.x * blockDim.x + threadIdx.x;
  if (gid >= N_NODES * HEADS_) return;
  int n = gid >> 2, head = gid & 3;
  const float* hp  = h + (((size_t)t * N_NODES + n) * HD_) + head * HID_;
  const float* alp = attn_l + (t * HEADS_ + head) * HID_;
  const float* arp = attn_r + (t * HEADS_ + head) * HID_;
  float sl = 0.f, sr = 0.f;
  #pragma unroll
  for (int i = 0; i < HID_; ++i) {
    float v = hp[i];
    sl = fmaf(v, alp[i], sl);
    sr = fmaf(v, arp[i], sr);
  }
  el[(t * N_NODES + n) * HEADS_ + head] = sl;
  er[(t * N_NODES + n) * HEADS_ + head] = sr;
}

__device__ __forceinline__ void atomicMaxF(float* addr, float val) {
  unsigned* ua  = (unsigned*)addr;
  unsigned  old = *ua;
  while (__uint_as_float(old) < val) {
    unsigned assumed = old;
    old = atomicCAS(ua, assumed, __float_as_uint(val));
    if (old == assumed) break;
  }
}

__device__ __forceinline__ void edge_sd(const int* src, const int* dst, int t, int e,
                                        int& s, int& d) {
  if (e < E_) {
    s = src[(size_t)t * E_ + e];
    d = dst[(size_t)t * E_ + e];
  } else {
    s = d = e - E_;   // self loop
  }
}

// ---------------- pass 1: segment max of leaky_relu(el[s]+er[d]) over dst ----------------
__global__ void edge_max_kernel(const int* __restrict__ src, const int* __restrict__ dst,
                                const float* __restrict__ el, const float* __restrict__ er,
                                float* __restrict__ emax, int t) {
  int gid = blockIdx.x * blockDim.x + threadIdx.x;
  if (gid >= EP_ * HEADS_) return;
  int e = gid >> 2, head = gid & 3;
  int s, d; edge_sd(src, dst, t, e, s, d);
  float x = el[(t * N_NODES + s) * HEADS_ + head] + er[(t * N_NODES + d) * HEADS_ + head];
  x = (x > 0.f) ? x : 0.2f * x;
  atomicMaxF(&emax[(t * N_NODES + d) * HEADS_ + head], x);
}

// ---------------- pass 2: denom = segment_sum(exp(e - emax[d])) ----------------
__global__ void edge_sum_kernel(const int* __restrict__ src, const int* __restrict__ dst,
                                const float* __restrict__ el, const float* __restrict__ er,
                                const float* __restrict__ emax, float* __restrict__ denom,
                                int t) {
  int gid = blockIdx.x * blockDim.x + threadIdx.x;
  if (gid >= EP_ * HEADS_) return;
  int e = gid >> 2, head = gid & 3;
  int s, d; edge_sd(src, dst, t, e, s, d);
  float x = el[(t * N_NODES + s) * HEADS_ + head] + er[(t * N_NODES + d) * HEADS_ + head];
  x = (x > 0.f) ? x : 0.2f * x;
  float ee = expf(x - emax[(t * N_NODES + d) * HEADS_ + head]);
  atomicAdd(&denom[(t * N_NODES + d) * HEADS_ + head], ee);
}

// ---------------- pass 3: rst[d] += a * h[s]   (one thread per (edge, feature)) ----------------
__global__ void edge_acc_kernel(const int* __restrict__ src, const int* __restrict__ dst,
                                const float* __restrict__ el, const float* __restrict__ er,
                                const float* __restrict__ emax, const float* __restrict__ denom,
                                const float* __restrict__ h, float* __restrict__ rst, int t) {
  int gid = blockIdx.x * blockDim.x + threadIdx.x;
  if (gid >= EP_ * HD_) return;
  int e = gid >> 6, f = gid & 63, head = f >> 4;
  int s, d; edge_sd(src, dst, t, e, s, d);
  int di = (t * N_NODES + d) * HEADS_ + head;
  float x = el[(t * N_NODES + s) * HEADS_ + head] + er[di];
  x = (x > 0.f) ? x : 0.2f * x;
  float a  = expf(x - emax[di]) / denom[di];
  float hv = h[((size_t)t * N_NODES + s) * HD_ + f];
  atomicAdd(&rst[((size_t)t * N_NODES + d) * HD_ + f], a * hv);
}

// ---------------- final: out[o] += dot(h_all, W_out[o])  (streaming GEMV) ----------------
#define GEMV_ITERS 32
__global__ void out_gemv_kernel(const float* __restrict__ hall,
                                const float* __restrict__ Wout,
                                float* __restrict__ out) {
  const int tid = threadIdx.x;
  float acc[OUT_] = {};
  long long base = (long long)blockIdx.x * (256 * GEMV_ITERS);
  for (int it = 0; it < GEMV_ITERS; ++it) {
    long long k = base + (long long)it * 256 + tid;
    if (k < (long long)KTOT) {
      float hv = hall[k];
      __builtin_prefetch(&hall[k + 4096], 0, 0);          // global_prefetch_b8
      __builtin_prefetch(&Wout[(size_t)k + 4096], 0, 0);  // lead stream o=0
      #pragma unroll
      for (int o = 0; o < OUT_; ++o)
        acc[o] = fmaf(hv, Wout[(size_t)o * KTOT + k], acc[o]);
    }
  }
  // wave32 reduction then cross-wave via LDS
  #pragma unroll
  for (int o = 0; o < OUT_; ++o)
    #pragma unroll
    for (int off = 16; off > 0; off >>= 1)
      acc[o] += __shfl_xor(acc[o], off, 32);

  __shared__ float sred[8][OUT_];
  int wave = tid >> 5, lane = tid & 31;
  if (lane == 0)
    #pragma unroll
    for (int o = 0; o < OUT_; ++o) sred[wave][o] = acc[o];
  __syncthreads();
  if (tid < OUT_) {
    float s = 0.f;
    #pragma unroll
    for (int w = 0; w < 8; ++w) s += sred[w][tid];
    atomicAdd(&out[tid], s);
  }
}

// ---------------- host launcher ----------------
extern "C" void kernel_launch(void* const* d_in, const int* in_sizes, int n_in,
                              void* d_out, int out_size, void* d_ws, size_t ws_size,
                              hipStream_t stream) {
  const float* node_feat = (const float*)d_in[0];
  const int*   src       = (const int*)d_in[1];
  const int*   dst       = (const int*)d_in[2];
  const float* W         = (const float*)d_in[3];
  const float* attn_l    = (const float*)d_in[4];
  const float* attn_r    = (const float*)d_in[5];
  const float* bias_conv = (const float*)d_in[6];
  const float* W_out     = (const float*)d_in[7];
  const float* b_out     = (const float*)d_in[8];
  float*       out       = (float*)d_out;

  // workspace carve (256B aligned)
  char* p = (char*)d_ws;
  auto carve = [&](size_t bytes) { char* r = p; p += (bytes + 255) & ~(size_t)255; return (void*)r; };
  float*    rst   = (float*)carve((size_t)KTOT * 4);                 // 25.6 MB (== h_all)
  float*    hbuf  = (float*)carve((size_t)KTOT * 4);                 // 25.6 MB
  float*    el    = (float*)carve((size_t)T_ * N_NODES * HEADS_ * 4);
  float*    er    = (float*)carve((size_t)T_ * N_NODES * HEADS_ * 4);
  float*    emax  = (float*)carve((size_t)T_ * N_NODES * HEADS_ * 4);
  float*    denom = (float*)carve((size_t)T_ * N_NODES * HEADS_ * 4);
  _Float16* xf16  = (_Float16*)carve((size_t)N_NODES * IN_F * 2);    // 25.6 MB
  _Float16* wt16  = (_Float16*)carve((size_t)T_ * HD_ * IN_F * 2);

  const int B = 256;
  gat_init_kernel<<<(KTOT + B - 1) / B, B, 0, stream>>>(rst, emax, denom, out, bias_conv, b_out);
  cvt_x_kernel<<<(N_NODES * IN_F + B - 1) / B, B, 0, stream>>>(node_feat, xf16);
  cvt_w_kernel<<<(T_ * HD_ * IN_F + B - 1) / B, B, 0, stream>>>(W, wt16);

  for (int t = 0; t < T_; ++t)
    gemm_wmma_kernel<<<(N_NODES + 127) / 128, 256, 0, stream>>>(xf16, wt16, hbuf, t);
  for (int t = 0; t < T_; ++t)
    elr_kernel<<<(N_NODES * HEADS_ + B - 1) / B, B, 0, stream>>>(hbuf, attn_l, attn_r, el, er, t);
  for (int t = 0; t < T_; ++t)
    edge_max_kernel<<<(EP_ * HEADS_ + B - 1) / B, B, 0, stream>>>(src, dst, el, er, emax, t);
  for (int t = 0; t < T_; ++t)
    edge_sum_kernel<<<(EP_ * HEADS_ + B - 1) / B, B, 0, stream>>>(src, dst, el, er, emax, denom, t);
  for (int t = 0; t < T_; ++t)
    edge_acc_kernel<<<(EP_ * HD_ + B - 1) / B, B, 0, stream>>>(src, dst, el, er, emax, denom,
                                                               hbuf, rst, t);
  int gemvBlocks = (KTOT + 256 * GEMV_ITERS - 1) / (256 * GEMV_ITERS);
  out_gemv_kernel<<<gemvBlocks, 256, 0, stream>>>(rst, W_out, out);
}
